// SE3AttentionUpdate_20607253086907
// MI455X (gfx1250) — matrix operand order repaired and checked by reading, not compiled, not run on hardware
//
#include <hip/hip_runtime.h>
#include <hip/hip_bf16.h>

#define N_NODES 10000
#define N_EDGES 320000
#define DIM 64
#define SDIM 9
#define HE 64
#define HEADS 8
#define OK 128
#define OV 128
#define KTP 576   // S*D

typedef __attribute__((ext_vector_type(16))) __bf16 v16bf;
typedef __attribute__((ext_vector_type(8)))  __bf16 v8bf;
typedef __attribute__((ext_vector_type(8)))  float  v8f;

#define WMMA_BF16(a, b, c) \
  __builtin_amdgcn_wmma_f32_16x16x32_bf16(false, (a), false, (b), (short)0, (c), false, false)

// ---- fragment loaders -------------------------------------------------------
// B fragment: Wt stored transposed [Ntot][K] bf16, lane column n=l&15, group g=l>>4,
// lo = [kb+8g, +8), hi = [kb+16+8g, +8) (16B aligned).
__device__ __forceinline__ v16bf bfrag(const __bf16* __restrict__ Wt, int K, int n, int kb, int g) {
  union { v16bf v; v8bf h[2]; } u;
  const __bf16* p = Wt + (size_t)n * K + kb;
  u.h[0] = *(const v8bf*)(p + 8 * g);
  u.h[1] = *(const v8bf*)(p + 16 + 8 * g);
  return u.v;
}
// A fragment from bf16 row (LDS): row m=l&15; lo = [kb+8g, +8), hi = [kb+16+8g, +8)
__device__ __forceinline__ v16bf afrag_b16(const __bf16* row, int kb, int g) {
  union { v16bf v; v8bf h[2]; } u;
  u.h[0] = *(const v8bf*)(row + kb + 8 * g);
  u.h[1] = *(const v8bf*)(row + kb + 16 + 8 * g);
  return u.v;
}
// A fragment from fp32 row in global, converting to bf16
__device__ __forceinline__ v16bf afrag_f32(const float* __restrict__ row, int kb, int g) {
  union { v16bf v; __bf16 e[16]; } u;
  const float* p0 = row + kb + 8 * g;
  const float* p1 = row + kb + 16 + 8 * g;
#pragma unroll
  for (int i = 0; i < 8; ++i) { u.e[i] = (__bf16)p0[i]; u.e[8 + i] = (__bf16)p1[i]; }
  return u.v;
}
// Tensor-product A fragment built on the fly: y[m, c] = sh[m, c>>6] * x[m, c&63].
// Each 8-run is 8-aligned within a 32-aligned kb block, so it never crosses a
// 64-boundary: a single sh scalar and a contiguous 8-float x run per half.
__device__ __forceinline__ v16bf afrag_tp(const float* xrow, const float* shrow, int kb, int g) {
  union { v16bf v; __bf16 e[16]; } u;
  int c0 = kb + 8 * g;
  int c1 = kb + 16 + 8 * g;
  float s0 = shrow[c0 >> 6];
  float s1 = shrow[c1 >> 6];
  const float* p0 = xrow + (c0 & 63);
  const float* p1 = xrow + (c1 & 63);
#pragma unroll
  for (int i = 0; i < 8; ++i) {
    u.e[i]     = (__bf16)(s0 * p0[i]);
    u.e[8 + i] = (__bf16)(s1 * p1[i]);
  }
  return u.v;
}

// ---- K1: weight prep (fp32 -> bf16 transposed) + zero accumulators ----------
__global__ void prep_kern(const float* __restrict__ Wq,
                          const float* __restrict__ kw1, const float* __restrict__ kw2,
                          const float* __restrict__ vw1, const float* __restrict__ vw2,
                          const float* __restrict__ Cqk, const float* __restrict__ Cv,
                          __bf16* wqt, __bf16* kw1t, __bf16* kw2t, __bf16* vw1t, __bf16* vw2t,
                          __bf16* cqkt, __bf16* cvt,
                          float* den, float* upd, float* ssum, float* ssq) {
  int i0 = blockIdx.x * blockDim.x + threadIdx.x;
  int stride = gridDim.x * blockDim.x;
  for (int i = i0; i < DIM * OK; i += stride) {        // Wq [64,128] -> wqt[o][d]
    int o = i & 127, d = i >> 7;
    wqt[o * DIM + d] = (__bf16)Wq[d * OK + o];
  }
  for (int i = i0; i < HE * HE; i += stride) {         // w1 [64,64] -> w1t[o][d]
    int o = i & 63, d = i >> 6;
    kw1t[o * HE + d] = (__bf16)kw1[d * HE + o];
    vw1t[o * HE + d] = (__bf16)vw1[d * HE + o];
  }
  for (int i = i0; i < HE * OK; i += stride) {         // w2 [64,128] -> w2t[o][d]
    int o = i & 127, d = i >> 7;
    kw2t[o * HE + d] = (__bf16)kw2[d * OK + o];
    vw2t[o * HE + d] = (__bf16)vw2[d * OK + o];
  }
  for (int i = i0; i < OK * KTP; i += stride) {        // C[d,s,o] -> Ct[o][s*64+d]
    int o = i / KTP, rem = i - o * KTP, s = rem >> 6, d = rem & 63;
    cqkt[o * KTP + s * DIM + d] = (__bf16)Cqk[(d * SDIM + s) * OK + o];
    cvt [o * KTP + s * DIM + d] = (__bf16)Cv [(d * SDIM + s) * OV + o];
  }
  for (int i = i0; i < N_NODES * HEADS; i += stride) den[i] = 0.0f;
  for (int i = i0; i < N_NODES * OV;    i += stride) upd[i] = 0.0f;
  if (i0 < DIM) { ssum[i0] = 0.0f; ssq[i0] = 0.0f; }
}

// ---- K2: q = tfn @ Wq (bf16 WMMA) ------------------------------------------
__global__ void __launch_bounds__(128) q_kern(const float* __restrict__ tfn,
                                              const __bf16* __restrict__ wqt,
                                              float* __restrict__ q) {
  int wave = (blockIdx.x * blockDim.x + threadIdx.x) >> 5;
  int lane = threadIdx.x & 31;
  if (wave >= N_NODES / 16) return;          // wave-uniform: EXEC all-ones inside
  int nl = lane & 15, g = lane >> 4;
  const float* row = tfn + (size_t)(wave * 16 + nl) * DIM;
  v16bf a0 = afrag_f32(row, 0, g);
  v16bf a1 = afrag_f32(row, 32, g);
#pragma unroll
  for (int j = 0; j < 8; ++j) {
    v8f c = {};
    c = WMMA_BF16(a0, bfrag(wqt, DIM, 16 * j + nl, 0, g), c);
    c = WMMA_BF16(a1, bfrag(wqt, DIM, 16 * j + nl, 32, g), c);
#pragma unroll
    for (int r = 0; r < 8; ++r)
      q[(size_t)(wave * 16 + 8 * g + r) * OK + 16 * j + nl] = c[r];
  }
}

// ---- edge MLP: wb = relu(ef@w1t+b1)@w2t+b2 (all WMMA) ----------------------
__device__ __forceinline__ void mlp_to_wb(v16bf aE0, v16bf aE1,
                                          const __bf16* __restrict__ w1t, const float* __restrict__ b1,
                                          const __bf16* __restrict__ w2t, const float* __restrict__ b2,
                                          __bf16 (*hb)[HE], float (*wb)[OK], int g, int nl) {
#pragma unroll
  for (int j = 0; j < 4; ++j) {
    v8f c = {};
    c = WMMA_BF16(aE0, bfrag(w1t, HE, 16 * j + nl, 0, g), c);
    c = WMMA_BF16(aE1, bfrag(w1t, HE, 16 * j + nl, 32, g), c);
    float bias = b1[16 * j + nl];
#pragma unroll
    for (int r = 0; r < 8; ++r)
      hb[8 * g + r][16 * j + nl] = (__bf16)fmaxf(c[r] + bias, 0.0f);
  }
  __syncthreads();
  v16bf ah0 = afrag_b16(&hb[nl][0], 0, g);
  v16bf ah1 = afrag_b16(&hb[nl][0], 32, g);
#pragma unroll
  for (int j = 0; j < 8; ++j) {
    v8f c = {};
    c = WMMA_BF16(ah0, bfrag(w2t, HE, 16 * j + nl, 0, g), c);
    c = WMMA_BF16(ah1, bfrag(w2t, HE, 16 * j + nl, 32, g), c);
    float bias = b2[16 * j + nl];
#pragma unroll
    for (int r = 0; r < 8; ++r)
      wb[8 * g + r][16 * j + nl] = c[r] + bias;
  }
  __syncthreads();
}

// ---- tensor product GEMM: acc[j] = (sh .* x)(16x576) @ Ct tiles (144 WMMA) --
__device__ __forceinline__ void tp_acc(const float* xrow, const float* shrow,
                                       const __bf16* __restrict__ ct,
                                       int g, int nl, v8f acc[8]) {
  v8f z = {};
#pragma unroll
  for (int j = 0; j < 8; ++j) acc[j] = z;
  for (int t = 0; t < 18; ++t) {
    v16bf a = afrag_tp(xrow, shrow, 32 * t, g);
#pragma unroll
    for (int j = 0; j < 8; ++j)
      acc[j] = WMMA_BF16(a, bfrag(ct, KTP, 16 * j + nl, 32 * t, g), acc[j]);
  }
}

// ---- K3: fused per-edge kernel (one wave = 16 edges, ~15.5KB LDS) ----------
__global__ void __launch_bounds__(32) edge_kern(
    const float* __restrict__ tfn, const float* __restrict__ ef, const float* __restrict__ sh,
    const int* __restrict__ eidx,
    const __bf16* __restrict__ kw1t, const float* __restrict__ kb1,
    const __bf16* __restrict__ kw2t, const float* __restrict__ kb2,
    const __bf16* __restrict__ vw1t, const float* __restrict__ vb1,
    const __bf16* __restrict__ vw2t, const float* __restrict__ vb2,
    const __bf16* __restrict__ cqkt, const __bf16* __restrict__ cvt,
    const float* __restrict__ q, float* __restrict__ den, float* __restrict__ upd) {
  __shared__ int sdst[16], ssrc[16];
  __shared__ __align__(16) float xs[16][DIM];
  __shared__ float shs[16][SDIM];
  __shared__ __align__(32) __bf16 hb[16][HE];
  __shared__ float wb[16][OK];
  __shared__ float pb[16][HEADS];

  int lane = threadIdx.x;
  int nl = lane & 15, g = lane >> 4;
  int e0 = blockIdx.x * 16;

  if (lane < 16) {
    sdst[lane] = eidx[e0 + lane];
    ssrc[lane] = eidx[N_EDGES + e0 + lane];
  }
  __syncthreads();

  // edge_features A fragments (fp32 -> bf16)
  const float* efrow = ef + (size_t)(e0 + nl) * HE;
  v16bf aE0 = afrag_f32(efrow, 0, g);
  v16bf aE1 = afrag_f32(efrow, 32, g);

  // gather x_dst rows and sh rows into LDS (fp32; TP A-frags built on the fly)
  for (int i = lane; i < 16 * DIM; i += 32) {
    int m = i >> 6, d = i & 63;
    xs[m][d] = tfn[(size_t)sdst[m] * DIM + d];
  }
  for (int i = lane; i < 16 * SDIM; i += 32) {
    int m = i / SDIM, s = i - m * SDIM;
    shs[m][s] = sh[(size_t)(e0 + m) * SDIM + s];
  }
  __syncthreads();

  // ---------------- K path ----------------
  mlp_to_wb(aE0, aE1, kw1t, kb1, kw2t, kb2, hb, wb, g, nl);    // wb = wk
  v8f acc[8];
  tp_acc(&xs[nl][0], &shs[nl][0], cqkt, g, nl, acc);           // acc = k_pre
#pragma unroll
  for (int j = 0; j < 8; ++j)
#pragma unroll
    for (int r = 0; r < 8; ++r) {
      int mr = 8 * g + r, col = 16 * j + nl;
      wb[mr][col] = acc[j][r] * wb[mr][col];                   // wb = k (in place, 1:1)
    }
  __syncthreads();

  // attention logits + exp (+ atomic denom); 16 edges x 8 heads = 4 passes
#pragma unroll
  for (int it = 0; it < 4; ++it) {
    int item = lane + 32 * it;
    int e = item >> 3, h = item & 7;
    const float* qrow = q + (size_t)ssrc[e] * OK + h * 16;
    float s = 0.0f;
#pragma unroll
    for (int d = 0; d < 16; ++d) s += qrow[d] * wb[e][h * 16 + d];
    float p = __expf(s);
    pb[e][h] = p;
    atomicAdd(&den[(size_t)ssrc[e] * HEADS + h], p);
  }
  __syncthreads();

  // ---------------- V path ----------------
  mlp_to_wb(aE0, aE1, vw1t, vb1, vw2t, vb2, hb, wb, g, nl);    // wb = wv
  tp_acc(&xs[nl][0], &shs[nl][0], cvt, g, nl, acc);            // acc = v_pre
#pragma unroll
  for (int j = 0; j < 8; ++j)
#pragma unroll
    for (int r = 0; r < 8; ++r) {
      int mr = 8 * g + r, col = 16 * j + nl;                   // head = col>>4 = j
      float val = acc[j][r] * wb[mr][col] * pb[mr][j];
      atomicAdd(&upd[(size_t)ssrc[mr] * OV + col], val);
    }
}

// ---- K4: per-node finish: W_out fuse + W_self + stats ----------------------
__global__ void __launch_bounds__(64) node_kern(const float* __restrict__ tfn,
                                                const float* __restrict__ upd,
                                                const float* __restrict__ den,
                                                const float* __restrict__ Wout,
                                                const float* __restrict__ Wself,
                                                float* __restrict__ outpre,
                                                float* __restrict__ ssum, float* __restrict__ ssq) {
  __shared__ float us[OV];
  int n = blockIdx.x, d = threadIdx.x;
  const float* ur = upd + (size_t)n * OV;
  const float* dr = den + (size_t)n * HEADS;
  us[d]      = ur[d]      / (dr[d >> 4]        + 1e-16f);
  us[d + 64] = ur[d + 64] / (dr[(d + 64) >> 4] + 1e-16f);
  __syncthreads();
  float acc = 0.0f;
  for (int c = 0; c < OV; ++c) acc += us[c] * Wout[c * DIM + d];
  const float* xr = tfn + (size_t)n * DIM;
  for (int k = 0; k < DIM; ++k) acc += xr[k] * Wself[k * DIM + d];
  outpre[(size_t)n * DIM + d] = acc;
  atomicAdd(&ssum[d], acc);
  atomicAdd(&ssq[d], acc * acc);
}

// ---- K5: batchnorm finalize ------------------------------------------------
__global__ void bn_kern(const float* __restrict__ outpre, const float* __restrict__ ssum,
                        const float* __restrict__ ssq, const float* __restrict__ gamma,
                        const float* __restrict__ beta, float* __restrict__ out) {
  int i = blockIdx.x * blockDim.x + threadIdx.x;
  if (i >= N_NODES * DIM) return;
  int d = i & 63;
  float mu = ssum[d] * (1.0f / N_NODES);
  float var = ssq[d] * (1.0f / N_NODES) - mu * mu;
  out[i] = (outpre[i] - mu) * rsqrtf(var + 1e-5f) * gamma[d] + beta[d];
}

// ---- launch ----------------------------------------------------------------
extern "C" void kernel_launch(void* const* d_in, const int* in_sizes, int n_in,
                              void* d_out, int out_size, void* d_ws, size_t ws_size,
                              hipStream_t stream) {
  const float* tfn  = (const float*)d_in[0];
  const float* ef   = (const float*)d_in[1];
  const float* sh   = (const float*)d_in[2];
  const float* Wq   = (const float*)d_in[3];
  const float* Cqk  = (const float*)d_in[4];
  const float* Cv   = (const float*)d_in[5];
  const float* kw1  = (const float*)d_in[6];
  const float* kb1  = (const float*)d_in[7];
  const float* kw2  = (const float*)d_in[8];
  const float* kb2  = (const float*)d_in[9];
  const float* vw1  = (const float*)d_in[10];
  const float* vb1  = (const float*)d_in[11];
  const float* vw2  = (const float*)d_in[12];
  const float* vb2  = (const float*)d_in[13];
  const float* Wout = (const float*)d_in[14];
  const float* Wself= (const float*)d_in[15];
  const float* gam  = (const float*)d_in[16];
  const float* bet  = (const float*)d_in[17];
  const int*   eidx = (const int*)d_in[18];
  float* out = (float*)d_out;

  char* ws = (char*)d_ws;
  size_t off = 0;
  auto take = [&](size_t bytes) { char* p = ws + off; off += (bytes + 255) & ~(size_t)255; return p; };
  __bf16* wqt  = (__bf16*)take(OK * DIM * 2);
  __bf16* kw1t = (__bf16*)take(HE * HE * 2);
  __bf16* kw2t = (__bf16*)take(OK * HE * 2);
  __bf16* vw1t = (__bf16*)take(HE * HE * 2);
  __bf16* vw2t = (__bf16*)take(OV * HE * 2);
  __bf16* cqkt = (__bf16*)take((size_t)OK * KTP * 2);
  __bf16* cvt  = (__bf16*)take((size_t)OV * KTP * 2);
  float* q     = (float*)take((size_t)N_NODES * OK * 4);
  float* den   = (float*)take((size_t)N_NODES * HEADS * 4);
  float* upd   = (float*)take((size_t)N_NODES * OV * 4);
  float* outpre= (float*)take((size_t)N_NODES * DIM * 4);
  float* ssum  = (float*)take(DIM * 4);
  float* ssq   = (float*)take(DIM * 4);

  prep_kern<<<512, 256, 0, stream>>>(Wq, kw1, kw2, vw1, vw2, Cqk, Cv,
                                     wqt, kw1t, kw2t, vw1t, vw2t, cqkt, cvt,
                                     den, upd, ssum, ssq);
  q_kern<<<(N_NODES / 16 * 32 + 127) / 128, 128, 0, stream>>>(tfn, wqt, q);
  edge_kern<<<N_EDGES / 16, 32, 0, stream>>>(tfn, ef, sh, eidx,
                                             kw1t, kb1, kw2t, kb2,
                                             vw1t, vb1, vw2t, vb2,
                                             cqkt, cvt, q, den, upd);
  node_kern<<<N_NODES, 64, 0, stream>>>(tfn, upd, den, Wout, Wself, outpre, ssum, ssq);
  bn_kern<<<(N_NODES * DIM + 255) / 256, 256, 0, stream>>>(outpre, ssum, ssq, gam, bet, out);
}
